// GATv2_JK_Model_43284680409187
// MI455X (gfx1250) — compile-verified, matrix operand order checked by reading, not run on hardware
//
#include <hip/hip_runtime.h>
#include <cstdint>
#include <cstddef>

// ---------------- problem constants (from reference) ----------------
#define N_NODES   10000
#define N_EDGES   120000
#define F_IN      64
#define C_DIM     128
#define H_HEADS   4
#define L_LAYERS  5
#define G_GRAPHS  500
#define OUT_DIM   1801
#define HL_DIM    320
#define GATE_DIM  (4 * HL_DIM)          // 1280
#define NH        (N_NODES * H_HEADS)   // 40000
#define EH        (N_EDGES * H_HEADS)   // 480000

// ---------------- WMMA types ----------------
typedef __attribute__((ext_vector_type(16))) _Float16 v16h;
typedef __attribute__((ext_vector_type(8)))  _Float16 v8h;
typedef __attribute__((ext_vector_type(8)))  float    v8f;

union F16Frag { v16h v16; v8h h2[2]; };

// ---------------- small device helpers ----------------
__device__ __forceinline__ float sigf(float x) { return 1.0f / (1.0f + __expf(-x)); }

__device__ __forceinline__ unsigned fkey(float f) {
    unsigned b = __float_as_uint(f);
    return (b & 0x80000000u) ? ~b : (b | 0x80000000u);
}
__device__ __forceinline__ float funkey(unsigned k) {
    return (k & 0x80000000u) ? __uint_as_float(k & 0x7fffffffu) : __uint_as_float(~k);
}

// CDNA5 async global->LDS copy (ASYNCcnt-tracked, bypasses VGPRs).
// ISA 15.18.3 op 98: VDST = per-lane LDS byte address, VADDR = global address.
__device__ __forceinline__ void async_b128(unsigned lds_off, const _Float16* gptr) {
    asm volatile("global_load_async_to_lds_b128 %0, %1, off"
                 :: "v"(lds_off), "v"(gptr) : "memory");
}
__device__ __forceinline__ void wait_async() {
    asm volatile("s_wait_asynccnt 0" ::: "memory");
}

// =====================================================================
// WMMA GEMM (all operands pre-converted f16):
//   C[M,N] = (accC ? C : 0) + A[M,K] * B^T (+bias) (relu?)   B stored [N,K]
//   optional dual f16 output C16 (row-major [M,N]) for chained GEMMs.
//   128 threads = 4 waves; macro tile 128(M) x 32(N); K stepped by 32.
//   Staging: double-buffered LDS filled by global_load_async_to_lds_b128;
//   next tile's async copies issue before current tile's WMMAs.
//   OOB rows/cols are address-clamped (EXEC stays all-ones for WMMA);
//   clamped garbage only reaches never-stored padding of C.
//   K must be a multiple of 32 (call sites: 64/128/320).
// =====================================================================
__global__ __launch_bounds__(128)
void gemm_wmma_f16(const _Float16* __restrict__ A, const _Float16* __restrict__ B,
                   const float* __restrict__ bias, float* __restrict__ C,
                   _Float16* __restrict__ C16,
                   int M, int N, int K, int accC, int doRelu)
{
    __shared__ _Float16 Al[2][128][40];  // 80B row stride: conflict-free frags
    __shared__ _Float16 Bl[2][32][40];   // [n][k]

    const int tid  = threadIdx.x;
    const int wave = tid >> 5;
    const int lane = tid & 31;
    const int m0 = blockIdx.x * 128;
    const int n0 = blockIdx.y * 32;
    const int r16 = lane & 15;
    const int kh  = (lane >> 4) * 8;   // K-half select per ISA 16-bit A/B layout

    v8f c00 = {}, c01 = {}, c10 = {}, c11 = {};

    // A tile: 128x32 halves = 512 b128 chunks -> 4/thread.
    // B tile: 32x32 halves = 128 b128 chunks -> 1/thread.
    auto issueTile = [&](int buf, int k0) {
#pragma unroll
        for (int j = 0; j < 4; ++j) {
            int i  = tid + 128 * j;
            int r  = i >> 2;              // tile row
            int c8 = (i & 3) * 8;         // half offset in row
            int gr = m0 + r;
            if (gr >= M) gr = M - 1;      // clamp, keep EXEC all-ones
            async_b128((unsigned)(size_t)&Al[buf][r][c8],
                       &A[(long)gr * K + k0 + c8]);
        }
        {
            int nn = tid >> 2;
            int k8 = (tid & 3) * 8;
            int gn = n0 + nn;
            if (gn >= N) gn = N - 1;
            async_b128((unsigned)(size_t)&Bl[buf][nn][k8],
                       &B[(long)gn * K + k0 + k8]);
        }
    };

    issueTile(0, 0);
    int buf = 0;
    for (int k0 = 0; k0 < K; k0 += 32) {
        wait_async();          // my copies into `buf` are complete
        __syncthreads();       // everyone's copies are complete
        if (k0 + 32 < K) issueTile(buf ^ 1, k0 + 32);   // overlap with WMMAs

        F16Frag a0, a1, b0, b1;
        const int ra = wave * 32 + r16;
        a0.h2[0] = *(const v8h*)&Al[buf][ra][kh];
        a0.h2[1] = *(const v8h*)&Al[buf][ra][16 + kh];
        a1.h2[0] = *(const v8h*)&Al[buf][ra + 16][kh];
        a1.h2[1] = *(const v8h*)&Al[buf][ra + 16][16 + kh];
        b0.h2[0] = *(const v8h*)&Bl[buf][r16][kh];
        b0.h2[1] = *(const v8h*)&Bl[buf][r16][16 + kh];
        b1.h2[0] = *(const v8h*)&Bl[buf][16 + r16][kh];
        b1.h2[1] = *(const v8h*)&Bl[buf][16 + r16][16 + kh];

        c00 = __builtin_amdgcn_wmma_f32_16x16x32_f16(false, a0.v16, false, b0.v16,
                                                     (short)0, c00, false, false);
        c01 = __builtin_amdgcn_wmma_f32_16x16x32_f16(false, a0.v16, false, b1.v16,
                                                     (short)0, c01, false, false);
        c10 = __builtin_amdgcn_wmma_f32_16x16x32_f16(false, a1.v16, false, b0.v16,
                                                     (short)0, c10, false, false);
        c11 = __builtin_amdgcn_wmma_f32_16x16x32_f16(false, a1.v16, false, b1.v16,
                                                     (short)0, c11, false, false);
        buf ^= 1;
    }

    // ---- epilogue: 16x16 f32 D layout (VGPR r -> M = r + (lane>=16)*8) ----
    const int colA = n0 + r16;
    const int colB = colA + 16;
    const int rOff = (lane >> 4) * 8;

    auto emit = [&](int row, int col, float v) {
        if (row < M && col < N) {
            float o = v;
            if (bias) o += bias[col];
            if (accC) o += C[(long)row * N + col];
            if (doRelu) o = fmaxf(o, 0.0f);
            C[(long)row * N + col] = o;
            if (C16) C16[(long)row * N + col] = (_Float16)o;
        }
    };
#pragma unroll
    for (int r = 0; r < 8; ++r) {
        int row0 = m0 + wave * 32 + rOff + r;
        int row1 = row0 + 16;
        emit(row0, colA, c00[r]);
        emit(row0, colB, c01[r]);
        emit(row1, colA, c10[r]);
        emit(row1, colB, c11[r]);
    }
}

// =====================================================================
// conversions / fills
// =====================================================================
__global__ void cvt_f16(_Float16* dst, const float* __restrict__ src, int n) {
    int i = blockIdx.x * 256 + threadIdx.x;
    if (i < n) dst[i] = (_Float16)src[i];
}
// src [K,N] fp32 -> dst [N,K] f16 (row-major), coalesced writes
__global__ void cvt_f16_t(_Float16* dst, const float* __restrict__ src, int K, int N) {
    int i = blockIdx.x * 256 + threadIdx.x;
    if (i < N * K) {
        int n = i / K, k = i - n * K;
        dst[i] = (_Float16)src[(long)k * N + n];
    }
}
__global__ void fill_f16(_Float16* p, float v, int n) {
    int i = blockIdx.x * 256 + threadIdx.x;
    if (i < n) p[i] = (_Float16)v;
}
__global__ void fill_f32(float* p, float v, int n) {
    int i = blockIdx.x * 256 + threadIdx.x;
    if (i < n) p[i] = v;
}
__global__ void fill_f32_ptr(float* p, const float* v, int n) {
    int i = blockIdx.x * 256 + threadIdx.x;
    if (i < n) p[i] = *v;
}
__global__ void fill_u32(unsigned* p, unsigned v, int n) {
    int i = blockIdx.x * 256 + threadIdx.x;
    if (i < n) p[i] = v;
}

// =====================================================================
// GATv2 edge passes
// =====================================================================
__global__ __launch_bounds__(256)
void gat_score_max(const float* __restrict__ xl, const float* __restrict__ xr,
                   const int* __restrict__ ei, const float* __restrict__ att,
                   float* __restrict__ score, unsigned* __restrict__ mkey)
{
    int lane = threadIdx.x & 31;
    int wid  = threadIdx.x >> 5;
    long eh = (long)blockIdx.x * 8 + wid;
    if (eh >= EH) return;
    int e  = (int)(eh >> 2);
    int hh = (int)(eh & 3);
    int src = ei[e], dst = ei[N_EDGES + e];
    float s = 0.0f;
    for (int c = lane; c < C_DIM; c += 32) {
        float v = xl[(long)src * (H_HEADS * C_DIM) + hh * C_DIM + c]
                + xr[(long)dst * (H_HEADS * C_DIM) + hh * C_DIM + c];
        v = (v > 0.0f) ? v : 0.2f * v;              // leaky_relu(0.2)
        s += v * att[hh * C_DIM + c];
    }
#pragma unroll
    for (int o = 16; o > 0; o >>= 1) s += __shfl_down(s, o, 32);
    if (lane == 0) {
        score[(long)e * H_HEADS + hh] = s;
        atomicMax(&mkey[(long)dst * H_HEADS + hh], fkey(s));
    }
}

__global__ void gat_max_fix(const unsigned* __restrict__ mkey, float* __restrict__ mval, int n) {
    int i = blockIdx.x * 256 + threadIdx.x;
    if (i >= n) return;
    float f = funkey(mkey[i]);
    mval[i] = isfinite(f) ? f : 0.0f;   // key==0 decodes to NaN -> no-edge case
}

__global__ void gat_exp_sum(const float* __restrict__ score, const int* __restrict__ ei,
                            const float* __restrict__ mval, float* __restrict__ exb,
                            float* __restrict__ denom)
{
    long i = (long)blockIdx.x * 256 + threadIdx.x;
    if (i >= EH) return;
    int e  = (int)(i >> 2);
    int hh = (int)(i & 3);
    int dst = ei[N_EDGES + e];
    float ex = __expf(score[i] - mval[(long)dst * H_HEADS + hh]);
    exb[i] = ex;
    atomicAdd(&denom[(long)dst * H_HEADS + hh], ex);
}

__global__ __launch_bounds__(128)
void gat_aggregate(const float* __restrict__ xl, const int* __restrict__ ei,
                   const float* __restrict__ exb, const float* __restrict__ denom,
                   float* __restrict__ acc)
{
    int e = blockIdx.x;
    int c = threadIdx.x;
    int src = ei[e], dst = ei[N_EDGES + e];
    __shared__ float al[H_HEADS];
    if (c < H_HEADS)
        al[c] = exb[(long)e * H_HEADS + c] / (denom[(long)dst * H_HEADS + c] + 1e-16f);
    __syncthreads();
    float s = 0.0f;
#pragma unroll
    for (int hh = 0; hh < H_HEADS; ++hh)
        s += xl[(long)src * (H_HEADS * C_DIM) + hh * C_DIM + c] * al[hh];
    atomicAdd(&acc[(long)dst * C_DIM + c], 0.25f * s);   // /H folded in
}

// bias + LayerNorm; dual-writes f16 copies feeding downstream GEMMs
__global__ __launch_bounds__(128)
void gat_bias_ln(const float* __restrict__ acc, const float* __restrict__ cb,
                 const float* __restrict__ lw, const float* __restrict__ lb,
                 float* __restrict__ seql,
                 _Float16* __restrict__ h16, _Float16* __restrict__ seq16l)
{
    int n = blockIdx.x, c = threadIdx.x;
    float v = acc[(long)n * C_DIM + c] + cb[c];
    float s = v, s2 = v * v;
#pragma unroll
    for (int o = 16; o > 0; o >>= 1) {
        s  += __shfl_down(s,  o, 32);
        s2 += __shfl_down(s2, o, 32);
    }
    __shared__ float ps[4], ps2[4], stat[2];
    int lane = c & 31, w = c >> 5;
    if (lane == 0) { ps[w] = s; ps2[w] = s2; }
    __syncthreads();
    if (c == 0) {
        float t  = ps[0] + ps[1] + ps[2] + ps[3];
        float t2 = ps2[0] + ps2[1] + ps2[2] + ps2[3];
        float mu  = t / (float)C_DIM;
        float var = t2 / (float)C_DIM - mu * mu;
        stat[0] = mu;
        stat[1] = rsqrtf(var + 1e-5f);
    }
    __syncthreads();
    float o = (v - stat[0]) * stat[1] * lw[c] + lb[c];
    seql[(long)n * C_DIM + c]   = o;             // fp32 for JK combine
    h16[(long)n * C_DIM + c]    = (_Float16)o;   // next layer's GEMM A
    seq16l[(long)n * C_DIM + c] = (_Float16)o;   // LSTM GEMM A
}

// =====================================================================
// LSTM pointwise with fused JK score accumulation; f16-only h output
// =====================================================================
__global__ __launch_bounds__(HL_DIM)
void lstm_pointwise(const float* __restrict__ gates,
                    const float* __restrict__ bih, const float* __restrict__ bhh,
                    float* __restrict__ c, _Float16* __restrict__ h16,
                    const float* __restrict__ jkw, float* __restrict__ scores, int tlog)
{
    int n = blockIdx.x, j = threadIdx.x;
    const float* g = gates + (long)n * GATE_DIM;
    float gi = g[j]              + bih[j]              + bhh[j];
    float gf = g[HL_DIM + j]     + bih[HL_DIM + j]     + bhh[HL_DIM + j];
    float gg = g[2 * HL_DIM + j] + bih[2 * HL_DIM + j] + bhh[2 * HL_DIM + j];
    float go = g[3 * HL_DIM + j] + bih[3 * HL_DIM + j] + bhh[3 * HL_DIM + j];
    float cv = sigf(gf) * c[(long)n * HL_DIM + j] + sigf(gi) * tanhf(gg);
    float hv = sigf(go) * tanhf(cv);
    c[(long)n * HL_DIM + j]   = cv;
    h16[(long)n * HL_DIM + j] = (_Float16)hv;    // next step's GEMM A

    float contrib = hv * jkw[j];
#pragma unroll
    for (int o = 16; o > 0; o >>= 1) contrib += __shfl_down(contrib, o, 32);
    __shared__ float part[HL_DIM / 32];
    int lane = j & 31, w = j >> 5;
    if (lane == 0) part[w] = contrib;
    __syncthreads();
    if (j == 0) {
        float t = 0.0f;
#pragma unroll
        for (int i = 0; i < HL_DIM / 32; ++i) t += part[i];
        scores[(long)tlog * N_NODES + n] += t;
    }
}

// =====================================================================
// JK softmax over L and weighted sum of seq
// =====================================================================
__global__ __launch_bounds__(128)
void jk_combine(const float* __restrict__ scores, const float* __restrict__ seq,
                float* __restrict__ node)
{
    int n = blockIdx.x, c = threadIdx.x;
    float sc[L_LAYERS];
    float m = -1e30f;
#pragma unroll
    for (int l = 0; l < L_LAYERS; ++l) {
        sc[l] = scores[(long)l * N_NODES + n];
        m = fmaxf(m, sc[l]);
    }
    float d = 0.0f;
#pragma unroll
    for (int l = 0; l < L_LAYERS; ++l) { sc[l] = __expf(sc[l] - m); d += sc[l]; }
    float inv = 1.0f / d;
    float o = 0.0f;
#pragma unroll
    for (int l = 0; l < L_LAYERS; ++l)
        o += sc[l] * inv * seq[((long)l * N_NODES + n) * C_DIM + c];
    node[(long)n * C_DIM + c] = o;
}

// =====================================================================
// graph mean-pool
// =====================================================================
__global__ __launch_bounds__(128)
void pool_accum(const float* __restrict__ node, const int* __restrict__ batch,
                float* __restrict__ pooled, float* __restrict__ cnt)
{
    int n = blockIdx.x, c = threadIdx.x;
    int g = batch[n];
    atomicAdd(&pooled[(long)g * C_DIM + c], node[(long)n * C_DIM + c]);
    if (c == 0) atomicAdd(&cnt[g], 1.0f);
}
__global__ void pool_div(float* pooled, _Float16* pooled16, const float* cnt) {
    int i = blockIdx.x * 256 + threadIdx.x;
    if (i < G_GRAPHS * C_DIM) {
        float v = pooled[i] / fmaxf(cnt[i >> 7], 1.0f);   // C=128 -> >>7
        pooled[i] = v;
        pooled16[i] = (_Float16)v;
    }
}

// =====================================================================
// host-side orchestration
// =====================================================================
static inline void launch_gemm(const _Float16* A, const _Float16* B, const float* bias,
                               float* C, _Float16* C16,
                               int M, int N, int K, int accC, int relu, hipStream_t s)
{
    dim3 grid((M + 127) / 128, (N + 31) / 32);
    gemm_wmma_f16<<<grid, dim3(128), 0, s>>>(A, B, bias, C, C16, M, N, K, accC, relu);
}

extern "C" void kernel_launch(void* const* d_in, const int* in_sizes, int n_in,
                              void* d_out, int out_size, void* d_ws, size_t ws_size,
                              hipStream_t stream)
{
    const float* x      = (const float*)d_in[0];
    const int*   ei     = (const int*)  d_in[1];
    const int*   batch  = (const int*)  d_in[2];
    const float* emb_W  = (const float*)d_in[3];
    const float* emb_b  = (const float*)d_in[4];
    const float* Wl     = (const float*)d_in[5];
    const float* bl     = (const float*)d_in[6];
    const float* Wr     = (const float*)d_in[7];
    const float* br     = (const float*)d_in[8];
    const float* att_a  = (const float*)d_in[9];
    const float* conv_b = (const float*)d_in[10];
    const float* ln_w   = (const float*)d_in[11];
    const float* ln_b   = (const float*)d_in[12];
    const float* Wih_f  = (const float*)d_in[13];
    const float* Whh_f  = (const float*)d_in[14];
    const float* bih_f  = (const float*)d_in[15];
    const float* bhh_f  = (const float*)d_in[16];
    const float* Wih_b  = (const float*)d_in[17];
    const float* Whh_b  = (const float*)d_in[18];
    const float* bih_b  = (const float*)d_in[19];
    const float* bhh_b  = (const float*)d_in[20];
    const float* jk_w   = (const float*)d_in[21];
    const float* jk_b   = (const float*)d_in[22];
    const float* m1_w   = (const float*)d_in[23];
    const float* m1_b   = (const float*)d_in[24];
    const float* m2_w   = (const float*)d_in[25];
    const float* m2_b   = (const float*)d_in[26];
    const float* res_w  = (const float*)d_in[27];
    const float* res_b  = (const float*)d_in[28];
    float* out = (float*)d_out;
    (void)in_sizes; (void)n_in; (void)out_size; (void)ws_size;

    // ---- bump-allocate workspace (deterministic layout) ----
    char* p = (char*)d_ws;
    auto bump = [&](size_t bytes) -> char* {
        char* r = p;
        p += (bytes + 255) & ~(size_t)255;
        return r;
    };
    // fp32 buffers
    float*    h_buf  = (float*)   bump(sizeof(float) * N_NODES * C_DIM);
    float*    xl     = (float*)   bump(sizeof(float) * N_NODES * H_HEADS * C_DIM);
    float*    xr     = (float*)   bump(sizeof(float) * N_NODES * H_HEADS * C_DIM);
    float*    score  = (float*)   bump(sizeof(float) * EH);
    float*    exb    = (float*)   bump(sizeof(float) * EH);
    unsigned* mkey   = (unsigned*)bump(sizeof(unsigned) * NH);
    float*    mval   = (float*)   bump(sizeof(float) * NH);
    float*    denom  = (float*)   bump(sizeof(float) * NH);
    float*    accb   = (float*)   bump(sizeof(float) * N_NODES * C_DIM);
    float*    seq    = (float*)   bump(sizeof(float) * L_LAYERS * N_NODES * C_DIM);
    float*    gates  = (float*)   bump(sizeof(float) * N_NODES * GATE_DIM);
    float*    cstate = (float*)   bump(sizeof(float) * N_NODES * HL_DIM);
    float*    scores = (float*)   bump(sizeof(float) * L_LAYERS * N_NODES);
    float*    node   = (float*)   bump(sizeof(float) * N_NODES * C_DIM);
    float*    pooled = (float*)   bump(sizeof(float) * G_GRAPHS * C_DIM);
    float*    cnt    = (float*)   bump(sizeof(float) * G_GRAPHS);
    float*    hdn    = (float*)   bump(sizeof(float) * G_GRAPHS * C_DIM);
    // f16 operand buffers
    _Float16* x16    = (_Float16*)bump(sizeof(_Float16) * N_NODES * F_IN);
    _Float16* h16    = (_Float16*)bump(sizeof(_Float16) * N_NODES * C_DIM);
    _Float16* seq16  = (_Float16*)bump(sizeof(_Float16) * L_LAYERS * N_NODES * C_DIM);
    _Float16* hst16  = (_Float16*)bump(sizeof(_Float16) * N_NODES * HL_DIM);
    _Float16* pool16 = (_Float16*)bump(sizeof(_Float16) * G_GRAPHS * C_DIM);
    _Float16* hdn16  = (_Float16*)bump(sizeof(_Float16) * G_GRAPHS * C_DIM);
    // f16 weights, stored [N,K]
    _Float16* embWt  = (_Float16*)bump(sizeof(_Float16) * C_DIM * F_IN);
    _Float16* Wlt    = (_Float16*)bump(sizeof(_Float16) * L_LAYERS * (H_HEADS*C_DIM) * C_DIM);
    _Float16* Wrt    = (_Float16*)bump(sizeof(_Float16) * L_LAYERS * (H_HEADS*C_DIM) * C_DIM);
    _Float16* Wihf16 = (_Float16*)bump(sizeof(_Float16) * GATE_DIM * C_DIM);
    _Float16* Whhf16 = (_Float16*)bump(sizeof(_Float16) * GATE_DIM * HL_DIM);
    _Float16* Wihb16 = (_Float16*)bump(sizeof(_Float16) * GATE_DIM * C_DIM);
    _Float16* Whhb16 = (_Float16*)bump(sizeof(_Float16) * GATE_DIM * HL_DIM);
    _Float16* m1wt   = (_Float16*)bump(sizeof(_Float16) * C_DIM * C_DIM);
    _Float16* m2wt   = (_Float16*)bump(sizeof(_Float16) * OUT_DIM * C_DIM);
    _Float16* reswt  = (_Float16*)bump(sizeof(_Float16) * OUT_DIM * C_DIM);

    auto blocks = [](int n) { return (n + 255) / 256; };

    // ---- 0) one-time f16 conversions (weights transposed to [N,K]) ----
    cvt_f16<<<blocks(N_NODES * F_IN), 256, 0, stream>>>(x16, x, N_NODES * F_IN);
    cvt_f16_t<<<blocks(C_DIM * F_IN), 256, 0, stream>>>(embWt, emb_W, F_IN, C_DIM);
    for (int l = 0; l < L_LAYERS; ++l) {
        cvt_f16_t<<<blocks(512 * C_DIM), 256, 0, stream>>>(
            Wlt + (size_t)l * 512 * C_DIM, Wl + (size_t)l * C_DIM * 512, C_DIM, 512);
        cvt_f16_t<<<blocks(512 * C_DIM), 256, 0, stream>>>(
            Wrt + (size_t)l * 512 * C_DIM, Wr + (size_t)l * C_DIM * 512, C_DIM, 512);
    }
    cvt_f16<<<blocks(GATE_DIM * C_DIM),  256, 0, stream>>>(Wihf16, Wih_f, GATE_DIM * C_DIM);
    cvt_f16<<<blocks(GATE_DIM * HL_DIM), 256, 0, stream>>>(Whhf16, Whh_f, GATE_DIM * HL_DIM);
    cvt_f16<<<blocks(GATE_DIM * C_DIM),  256, 0, stream>>>(Wihb16, Wih_b, GATE_DIM * C_DIM);
    cvt_f16<<<blocks(GATE_DIM * HL_DIM), 256, 0, stream>>>(Whhb16, Whh_b, GATE_DIM * HL_DIM);
    cvt_f16_t<<<blocks(C_DIM * C_DIM),   256, 0, stream>>>(m1wt, m1_w, C_DIM, C_DIM);
    cvt_f16_t<<<blocks(OUT_DIM * C_DIM), 256, 0, stream>>>(m2wt, m2_w, C_DIM, OUT_DIM);
    cvt_f16_t<<<blocks(OUT_DIM * C_DIM), 256, 0, stream>>>(reswt, res_w, C_DIM, OUT_DIM);

    // ---- 1) embedding: h = x @ emb_W + emb_b (dual f16 out) ----
    launch_gemm(x16, embWt, emb_b, h_buf, h16, N_NODES, C_DIM, F_IN, 0, 0, stream);

    // ---- 2) GATv2 layers ----
    for (int l = 0; l < L_LAYERS; ++l) {
        launch_gemm(h16, Wlt + (size_t)l * 512 * C_DIM, bl + (size_t)l * 512,
                    xl, nullptr, N_NODES, 512, C_DIM, 0, 0, stream);
        launch_gemm(h16, Wrt + (size_t)l * 512 * C_DIM, br + (size_t)l * 512,
                    xr, nullptr, N_NODES, 512, C_DIM, 0, 0, stream);

        fill_u32<<<blocks(NH), 256, 0, stream>>>(mkey, 0u, NH);
        fill_f32<<<blocks(NH), 256, 0, stream>>>(denom, 0.0f, NH);
        fill_f32<<<blocks(N_NODES * C_DIM), 256, 0, stream>>>(accb, 0.0f, N_NODES * C_DIM);

        gat_score_max<<<(EH + 7) / 8, 256, 0, stream>>>(
            xl, xr, ei, att_a + (size_t)l * H_HEADS * C_DIM, score, mkey);
        gat_max_fix<<<blocks(NH), 256, 0, stream>>>(mkey, mval, NH);
        gat_exp_sum<<<blocks(EH), 256, 0, stream>>>(score, ei, mval, exb, denom);
        gat_aggregate<<<N_EDGES, 128, 0, stream>>>(xl, ei, exb, denom, accb);

        gat_bias_ln<<<N_NODES, 128, 0, stream>>>(
            accb, conv_b + (size_t)l * C_DIM, ln_w + (size_t)l * C_DIM,
            ln_b + (size_t)l * C_DIM,
            seq + (size_t)l * N_NODES * C_DIM,
            h16, seq16 + (size_t)l * N_NODES * C_DIM);
    }

    // ---- 3) bidirectional LSTM with fused JK score accumulation ----
    fill_f32_ptr<<<blocks(L_LAYERS * N_NODES), 256, 0, stream>>>(scores, jk_b,
                                                                 L_LAYERS * N_NODES);
    // forward
    fill_f16<<<blocks(N_NODES * HL_DIM), 256, 0, stream>>>(hst16, 0.0f, N_NODES * HL_DIM);
    fill_f32<<<blocks(N_NODES * HL_DIM), 256, 0, stream>>>(cstate, 0.0f, N_NODES * HL_DIM);
    for (int t = 0; t < L_LAYERS; ++t) {
        const _Float16* st = seq16 + (size_t)t * N_NODES * C_DIM;
        launch_gemm(st,    Wihf16, nullptr, gates, nullptr, N_NODES, GATE_DIM, C_DIM,  0, 0, stream);
        launch_gemm(hst16, Whhf16, nullptr, gates, nullptr, N_NODES, GATE_DIM, HL_DIM, 1, 0, stream);
        lstm_pointwise<<<N_NODES, HL_DIM, 0, stream>>>(gates, bih_f, bhh_f, cstate, hst16,
                                                       jk_w, scores, t);
    }
    // backward
    fill_f16<<<blocks(N_NODES * HL_DIM), 256, 0, stream>>>(hst16, 0.0f, N_NODES * HL_DIM);
    fill_f32<<<blocks(N_NODES * HL_DIM), 256, 0, stream>>>(cstate, 0.0f, N_NODES * HL_DIM);
    for (int t = 0; t < L_LAYERS; ++t) {
        int ts = L_LAYERS - 1 - t;
        const _Float16* st = seq16 + (size_t)ts * N_NODES * C_DIM;
        launch_gemm(st,    Wihb16, nullptr, gates, nullptr, N_NODES, GATE_DIM, C_DIM,  0, 0, stream);
        launch_gemm(hst16, Whhb16, nullptr, gates, nullptr, N_NODES, GATE_DIM, HL_DIM, 1, 0, stream);
        lstm_pointwise<<<N_NODES, HL_DIM, 0, stream>>>(gates, bih_b, bhh_b, cstate, hst16,
                                                       jk_w + HL_DIM, scores, ts);
    }

    // ---- 4) JK attention combine ----
    jk_combine<<<N_NODES, 128, 0, stream>>>(scores, seq, node);

    // ---- 5) graph mean pooling ----
    fill_f32<<<blocks(G_GRAPHS * C_DIM), 256, 0, stream>>>(pooled, 0.0f, G_GRAPHS * C_DIM);
    fill_f32<<<blocks(G_GRAPHS), 256, 0, stream>>>(cnt, 0.0f, G_GRAPHS);
    pool_accum<<<N_NODES, 128, 0, stream>>>(node, batch, pooled, cnt);
    pool_div<<<blocks(G_GRAPHS * C_DIM), 256, 0, stream>>>(pooled, pool16, cnt);

    // ---- 6) output MLP + residual head ----
    launch_gemm(pool16, m1wt, m1_b, hdn, hdn16, G_GRAPHS, C_DIM, C_DIM, 0, 1, stream);   // relu
    launch_gemm(hdn16,  m2wt, m2_b, out, nullptr, G_GRAPHS, OUT_DIM, C_DIM, 0, 0, stream);
    launch_gemm(pool16, reswt, res_b, out, nullptr, G_GRAPHS, OUT_DIM, C_DIM, 1, 0, stream); // +=
}
